// SentimentClassificationMoE_74603581931976
// MI455X (gfx1250) — matrix operand order repaired and checked by reading, not compiled, not run on hardware
//
#include <hip/hip_runtime.h>
#include <hip/hip_bf16.h>
#include <hip/hip_fp16.h>

typedef __attribute__((ext_vector_type(16))) _Float16 v16h;
typedef __attribute__((ext_vector_type(8)))  _Float16 v8h;
typedef __attribute__((ext_vector_type(8)))  float    v8f;

constexpr int Hdim = 1024;
constexpr int Fdim = 4096;
constexpr int Edim = 8;
constexpr int Sdim = 128;
constexpr int Ntok = 4096;

constexpr int BM = 128, BN = 128, BK = 32;
constexpr int LDT = 40;               // padded LDS stride in halves (80 B)
constexpr int TILES_M = Ntok / BM;    // 32

// ---------------------------------------------------------------------------
// 1) Embedding gather + mean pool over S.  h[n] = mean_s emb[x[n,s]]
// ---------------------------------------------------------------------------
__global__ __launch_bounds__(256) void k_embed(const int* __restrict__ x,
                                               const float* __restrict__ emb,
                                               float* __restrict__ hf32,
                                               _Float16* __restrict__ hf16)
{
    const int n = blockIdx.x;
    const int c = threadIdx.x * 4;                 // H=1024 = 256*4
    const int* xs = x + (size_t)n * Sdim;
    float a0 = 0.f, a1 = 0.f, a2 = 0.f, a3 = 0.f;
    for (int s = 0; s < Sdim; ++s) {
        const int id = xs[s];
        const float4 v = *(const float4*)(emb + (size_t)id * Hdim + c);
        a0 += v.x; a1 += v.y; a2 += v.z; a3 += v.w;
    }
    const float inv = 1.0f / (float)Sdim;
    a0 *= inv; a1 *= inv; a2 *= inv; a3 *= inv;
    float4 o; o.x = a0; o.y = a1; o.z = a2; o.w = a3;
    *(float4*)(hf32 + (size_t)n * Hdim + c) = o;
    union { _Float16 h[4]; uint2 u; } pk;
    pk.h[0] = (_Float16)a0; pk.h[1] = (_Float16)a1;
    pk.h[2] = (_Float16)a2; pk.h[3] = (_Float16)a3;
    *(uint2*)(hf16 + (size_t)n * Hdim + c) = pk.u;
}

__global__ void k_zero(int* __restrict__ counts)
{
    if (threadIdx.x < Edim) counts[threadIdx.x] = 0;
}

// ---------------------------------------------------------------------------
// 2) Top-1 gating
// ---------------------------------------------------------------------------
__global__ __launch_bounds__(256) void k_gate(const float* __restrict__ hf32,
                                              const float* __restrict__ Wg,
                                              float* __restrict__ gatev,
                                              int* __restrict__ counts,
                                              int* __restrict__ lists)
{
    const int n = blockIdx.x, tid = threadIdx.x;
    float part[Edim];
#pragma unroll
    for (int e = 0; e < Edim; ++e) part[e] = 0.f;
    const float* hr = hf32 + (size_t)n * Hdim;
    for (int c = tid; c < Hdim; c += 256) {
        const float hv = hr[c];
        const float* wr = Wg + (size_t)c * Edim;
#pragma unroll
        for (int e = 0; e < Edim; ++e) part[e] += hv * wr[e];
    }
#pragma unroll
    for (int e = 0; e < Edim; ++e)
        for (int off = 16; off; off >>= 1) part[e] += __shfl_xor(part[e], off, 32);
    __shared__ float sred[8][Edim];
    const int wave = tid >> 5, lane = tid & 31;
    if (lane == 0) {
#pragma unroll
        for (int e = 0; e < Edim; ++e) sred[wave][e] = part[e];
    }
    __syncthreads();
    if (tid == 0) {
        float lg[Edim];
#pragma unroll
        for (int e = 0; e < Edim; ++e) {
            float s = 0.f;
            for (int w = 0; w < 8; ++w) s += sred[w][e];
            lg[e] = s;
        }
        float mx = lg[0]; int bi = 0;
#pragma unroll
        for (int e = 1; e < Edim; ++e)
            if (lg[e] > mx) { mx = lg[e]; bi = e; }   // first-max like jnp.argmax
        float sum = 0.f;
#pragma unroll
        for (int e = 0; e < Edim; ++e) sum += expf(lg[e] - mx);
        gatev[n] = 1.0f / sum;                        // softmax prob of argmax
        const int pos = atomicAdd(&counts[bi], 1);
        lists[bi * Ntok + pos] = n;
    }
}

// ---------------------------------------------------------------------------
// 3) Expert FFN layer 1 (double-buffered, register-pipelined WMMA GEMM)
//    h1[tok] = relu(h[tok] @ W1[e] + b1[e])
// ---------------------------------------------------------------------------
__global__ __launch_bounds__(256) void k_ffn1(const _Float16* __restrict__ hf16,
                                              const float* __restrict__ W1,
                                              const float* __restrict__ b1,
                                              const int* __restrict__ lists,
                                              const int* __restrict__ counts,
                                              _Float16* __restrict__ h1)
{
    const int e    = blockIdx.x / TILES_M;
    const int row0 = (blockIdx.x % TILES_M) * BM;
    const int cnt  = counts[e];
    if (row0 >= cnt) return;
    const int n0 = blockIdx.y * BN;

    __shared__ _Float16 As[2][BM * LDT];
    __shared__ _Float16 Bs[2][BN * LDT];
    __shared__ int toks[BM];

    const int tid = threadIdx.x;
    if (tid < BM) {
        const int r = row0 + tid;
        toks[tid] = (r < cnt) ? lists[e * Ntok + r] : -1;
    }
    __syncthreads();

    const float* __restrict__ Be = W1 + (size_t)e * Hdim * Fdim;

    const int arow = tid >> 1;
    const int acol = (tid & 1) * 16;
    const int atok = toks[arow];
    const _Float16* aGlob = (atok >= 0) ? (hf16 + (size_t)atok * Hdim + acol) : nullptr;

    const int bk = (tid >> 4) * 2;       // two consecutive k rows
    const int bn = (tid & 15) * 8;       // 8 columns

    const int wave = tid >> 5, lane = tid & 31;
    const int wm = (wave >> 2) * 64;
    const int wn = (wave & 3) * 32;
    const int lhi = lane >> 4, lm = lane & 15;

    v8f acc[4][2] = {};

    uint4  ad0, ad1;                      // prefetched A (f16)
    float4 w0a, w0b, w1a, w1b;            // prefetched B rows (f32)

    auto globalLoad = [&](int k0) {
        ad0 = uint4{0, 0, 0, 0}; ad1 = uint4{0, 0, 0, 0};
        if (aGlob) {
            const uint4* p = (const uint4*)(aGlob + k0);
            ad0 = p[0]; ad1 = p[1];
        }
        const float* s0 = Be + (size_t)(k0 + bk) * Fdim + n0 + bn;
        const float* s1 = s0 + Fdim;
        w0a = *(const float4*)s0; w0b = *(const float4*)(s0 + 4);
        w1a = *(const float4*)s1; w1b = *(const float4*)(s1 + 4);
    };

    auto ldsStore = [&](int buf) {
        _Float16* A = &As[buf][0];
        _Float16* B = &Bs[buf][0];
        *(uint4*)(&A[arow * LDT + acol])     = ad0;
        *(uint4*)(&A[arow * LDT + acol + 8]) = ad1;
        const float c0[8] = {w0a.x, w0a.y, w0a.z, w0a.w, w0b.x, w0b.y, w0b.z, w0b.w};
        const float c1[8] = {w1a.x, w1a.y, w1a.z, w1a.w, w1b.x, w1b.y, w1b.z, w1b.w};
#pragma unroll
        for (int i = 0; i < 8; ++i) {
            union { _Float16 h[2]; unsigned u; } p;
            p.h[0] = (_Float16)c0[i];
            p.h[1] = (_Float16)c1[i];
            *(unsigned*)(&B[(bn + i) * LDT + bk]) = p.u;   // Bs[n][k] (k-transposed)
        }
    };

    auto compute = [&](int buf) {
        const _Float16* A = &As[buf][0];
        const _Float16* B = &Bs[buf][0];
        v16h af[4], bf[2];
#pragma unroll
        for (int mt = 0; mt < 4; ++mt) {
            const _Float16* ap = &A[(wm + mt * 16 + lm) * LDT + lhi * 8];
            const v8h lo = *(const v8h*)ap;
            const v8h hi = *(const v8h*)(ap + 16);
#pragma unroll
            for (int i = 0; i < 8; ++i) { af[mt][i] = lo[i]; af[mt][i + 8] = hi[i]; }
        }
#pragma unroll
        for (int nt = 0; nt < 2; ++nt) {
            const _Float16* bp = &B[(wn + nt * 16 + lm) * LDT + lhi * 8];
            const v8h lo = *(const v8h*)bp;
            const v8h hi = *(const v8h*)(bp + 16);
#pragma unroll
            for (int i = 0; i < 8; ++i) { bf[nt][i] = lo[i]; bf[nt][i + 8] = hi[i]; }
        }
#pragma unroll
        for (int mt = 0; mt < 4; ++mt)
#pragma unroll
            for (int nt = 0; nt < 2; ++nt)
                acc[mt][nt] = __builtin_amdgcn_wmma_f32_16x16x32_f16(
                    false, af[mt], false, bf[nt], (short)0, acc[mt][nt], false, false);
    };

    // pipelined main loop: one barrier per K-step, global loads overlap WMMA
    globalLoad(0);
    ldsStore(0);
    __syncthreads();
    constexpr int NK = Hdim / BK;
    for (int t = 0; t < NK; ++t) {
        const int buf = t & 1;
        if (t + 1 < NK) globalLoad((t + 1) * BK);
        compute(buf);
        if (t + 1 < NK) {
            ldsStore(buf ^ 1);
            __syncthreads();
        }
    }

    // ---- epilogue: bias + relu -> f16 scatter by token row ----
#pragma unroll
    for (int nt = 0; nt < 2; ++nt) {
        const int ncol = n0 + wn + nt * 16 + lm;
        const float bias = b1[e * Fdim + ncol];
#pragma unroll
        for (int mt = 0; mt < 4; ++mt) {
#pragma unroll
            for (int r = 0; r < 8; ++r) {
                const int m = wm + mt * 16 + 8 * lhi + r;
                const int tok = toks[m];
                if (tok >= 0) {
                    float v = acc[mt][nt][r] + bias;
                    v = v > 0.f ? v : 0.f;
                    h1[(size_t)tok * Fdim + ncol] = (_Float16)v;
                }
            }
        }
    }
}

// ---------------------------------------------------------------------------
// 4) Expert FFN layer 2 (same pipelined structure)
//    moe[tok] = gate[tok] * (h1[tok] @ W2[e] + b2[e])
// ---------------------------------------------------------------------------
__global__ __launch_bounds__(256) void k_ffn2(const _Float16* __restrict__ h1,
                                              const float* __restrict__ W2,
                                              const float* __restrict__ b2,
                                              const int* __restrict__ lists,
                                              const int* __restrict__ counts,
                                              const float* __restrict__ gatev,
                                              float* __restrict__ moe)
{
    const int e    = blockIdx.x / TILES_M;
    const int row0 = (blockIdx.x % TILES_M) * BM;
    const int cnt  = counts[e];
    if (row0 >= cnt) return;
    const int n0 = blockIdx.y * BN;

    __shared__ _Float16 As[2][BM * LDT];
    __shared__ _Float16 Bs[2][BN * LDT];
    __shared__ int toks[BM];
    __shared__ float gv[BM];

    const int tid = threadIdx.x;
    if (tid < BM) {
        const int r = row0 + tid;
        const int t = (r < cnt) ? lists[e * Ntok + r] : -1;
        toks[tid] = t;
        gv[tid]   = (t >= 0) ? gatev[t] : 0.f;
    }
    __syncthreads();

    const float* __restrict__ Be = W2 + (size_t)e * Fdim * Hdim;

    const int arow = tid >> 1;
    const int acol = (tid & 1) * 16;
    const int atok = toks[arow];
    const _Float16* aGlob = (atok >= 0) ? (h1 + (size_t)atok * Fdim + acol) : nullptr;

    const int bk = (tid >> 4) * 2;
    const int bn = (tid & 15) * 8;

    const int wave = tid >> 5, lane = tid & 31;
    const int wm = (wave >> 2) * 64;
    const int wn = (wave & 3) * 32;
    const int lhi = lane >> 4, lm = lane & 15;

    v8f acc[4][2] = {};

    uint4  ad0, ad1;
    float4 w0a, w0b, w1a, w1b;

    auto globalLoad = [&](int k0) {
        ad0 = uint4{0, 0, 0, 0}; ad1 = uint4{0, 0, 0, 0};
        if (aGlob) {
            const uint4* p = (const uint4*)(aGlob + k0);
            ad0 = p[0]; ad1 = p[1];
        }
        const float* s0 = Be + (size_t)(k0 + bk) * Hdim + n0 + bn;
        const float* s1 = s0 + Hdim;
        w0a = *(const float4*)s0; w0b = *(const float4*)(s0 + 4);
        w1a = *(const float4*)s1; w1b = *(const float4*)(s1 + 4);
    };

    auto ldsStore = [&](int buf) {
        _Float16* A = &As[buf][0];
        _Float16* B = &Bs[buf][0];
        *(uint4*)(&A[arow * LDT + acol])     = ad0;
        *(uint4*)(&A[arow * LDT + acol + 8]) = ad1;
        const float c0[8] = {w0a.x, w0a.y, w0a.z, w0a.w, w0b.x, w0b.y, w0b.z, w0b.w};
        const float c1[8] = {w1a.x, w1a.y, w1a.z, w1a.w, w1b.x, w1b.y, w1b.z, w1b.w};
#pragma unroll
        for (int i = 0; i < 8; ++i) {
            union { _Float16 h[2]; unsigned u; } p;
            p.h[0] = (_Float16)c0[i];
            p.h[1] = (_Float16)c1[i];
            *(unsigned*)(&B[(bn + i) * LDT + bk]) = p.u;
        }
    };

    auto compute = [&](int buf) {
        const _Float16* A = &As[buf][0];
        const _Float16* B = &Bs[buf][0];
        v16h af[4], bf[2];
#pragma unroll
        for (int mt = 0; mt < 4; ++mt) {
            const _Float16* ap = &A[(wm + mt * 16 + lm) * LDT + lhi * 8];
            const v8h lo = *(const v8h*)ap;
            const v8h hi = *(const v8h*)(ap + 16);
#pragma unroll
            for (int i = 0; i < 8; ++i) { af[mt][i] = lo[i]; af[mt][i + 8] = hi[i]; }
        }
#pragma unroll
        for (int nt = 0; nt < 2; ++nt) {
            const _Float16* bp = &B[(wn + nt * 16 + lm) * LDT + lhi * 8];
            const v8h lo = *(const v8h*)bp;
            const v8h hi = *(const v8h*)(bp + 16);
#pragma unroll
            for (int i = 0; i < 8; ++i) { bf[nt][i] = lo[i]; bf[nt][i + 8] = hi[i]; }
        }
#pragma unroll
        for (int mt = 0; mt < 4; ++mt)
#pragma unroll
            for (int nt = 0; nt < 2; ++nt)
                acc[mt][nt] = __builtin_amdgcn_wmma_f32_16x16x32_f16(
                    false, af[mt], false, bf[nt], (short)0, acc[mt][nt], false, false);
    };

    globalLoad(0);
    ldsStore(0);
    __syncthreads();
    constexpr int NK = Fdim / BK;
    for (int t = 0; t < NK; ++t) {
        const int buf = t & 1;
        if (t + 1 < NK) globalLoad((t + 1) * BK);
        compute(buf);
        if (t + 1 < NK) {
            ldsStore(buf ^ 1);
            __syncthreads();
        }
    }

#pragma unroll
    for (int nt = 0; nt < 2; ++nt) {
        const int ncol = n0 + wn + nt * 16 + lm;
        const float bias = b2[e * Hdim + ncol];
#pragma unroll
        for (int mt = 0; mt < 4; ++mt) {
#pragma unroll
            for (int r = 0; r < 8; ++r) {
                const int m = wm + mt * 16 + 8 * lhi + r;
                const int tok = toks[m];
                if (tok >= 0)
                    moe[(size_t)tok * Hdim + ncol] = gv[m] * (acc[mt][nt][r] + bias);
            }
        }
    }
}

// ---------------------------------------------------------------------------
// 5) Classifier + log_softmax over O=2
// ---------------------------------------------------------------------------
__global__ __launch_bounds__(128) void k_cls(const float* __restrict__ moe,
                                             const float* __restrict__ Wfc,
                                             const float* __restrict__ bfc,
                                             float* __restrict__ out)
{
    const int n = blockIdx.x, tid = threadIdx.x;
    float a0 = 0.f, a1 = 0.f;
    const float* r = moe + (size_t)n * Hdim;
    for (int c = tid; c < Hdim; c += 128) {
        const float v = r[c];
        a0 += v * Wfc[c * 2 + 0];
        a1 += v * Wfc[c * 2 + 1];
    }
    for (int off = 16; off; off >>= 1) {
        a0 += __shfl_xor(a0, off, 32);
        a1 += __shfl_xor(a1, off, 32);
    }
    __shared__ float s0[4], s1[4];
    const int wave = tid >> 5, lane = tid & 31;
    if (lane == 0) { s0[wave] = a0; s1[wave] = a1; }
    __syncthreads();
    if (tid == 0) {
        float z0 = bfc[0], z1 = bfc[1];
        for (int w = 0; w < 4; ++w) { z0 += s0[w]; z1 += s1[w]; }
        const float m = fmaxf(z0, z1);
        const float lse = m + logf(expf(z0 - m) + expf(z1 - m));
        out[(size_t)n * 2 + 0] = z0 - lse;
        out[(size_t)n * 2 + 1] = z1 - lse;
    }
}

// ---------------------------------------------------------------------------
extern "C" void kernel_launch(void* const* d_in, const int* in_sizes, int n_in,
                              void* d_out, int out_size, void* d_ws, size_t ws_size,
                              hipStream_t stream)
{
    (void)in_sizes; (void)n_in; (void)out_size; (void)ws_size;
    const int*   x   = (const int*)d_in[0];
    const float* emb = (const float*)d_in[1];
    const float* Wg  = (const float*)d_in[2];
    const float* W1  = (const float*)d_in[3];
    const float* b1  = (const float*)d_in[4];
    const float* W2  = (const float*)d_in[5];
    const float* b2  = (const float*)d_in[6];
    const float* Wfc = (const float*)d_in[7];
    const float* bfc = (const float*)d_in[8];
    float* out = (float*)d_out;

    char* p = (char*)d_ws;
    auto alloc = [&](size_t bytes) {
        char* r = p;
        p += (bytes + 255) & ~size_t(255);
        return r;
    };
    float*    hf32   = (float*)   alloc((size_t)Ntok * Hdim * 4);
    _Float16* hf16   = (_Float16*)alloc((size_t)Ntok * Hdim * 2);
    float*    gatev  = (float*)   alloc((size_t)Ntok * 4);
    int*      counts = (int*)     alloc((size_t)Edim * 4);
    int*      lists  = (int*)     alloc((size_t)Edim * Ntok * 4);
    _Float16* h1     = (_Float16*)alloc((size_t)Ntok * Fdim * 2);
    float*    moe    = (float*)   alloc((size_t)Ntok * Hdim * 4);

    k_embed<<<Ntok, 256, 0, stream>>>(x, emb, hf32, hf16);
    k_zero <<<1, 32, 0, stream>>>(counts);
    k_gate <<<Ntok, 256, 0, stream>>>(hf32, Wg, gatev, counts, lists);
    k_ffn1 <<<dim3(Edim * TILES_M, Fdim / BN), 256, 0, stream>>>(hf16, W1, b1, lists, counts, h1);
    k_ffn2 <<<dim3(Edim * TILES_M, Hdim / BN), 256, 0, stream>>>(h1, W2, b2, lists, counts, gatev, moe);
    k_cls  <<<Ntok, 128, 0, stream>>>(moe, Wfc, bfc, out);
}
